// EnhancedEdgeAwareGNN_30734785970664
// MI455X (gfx1250) — matrix operand order repaired and checked by reading, not compiled, not run on hardware
//
#include <hip/hip_runtime.h>
#include <hip/hip_bf16.h>
#include <stdint.h>

#define NN 10000
#define NE 160000
#define NET 170000
#define ND 8
#define ED 4
#define HDIM 128
#define OUTD 256
#define NL 6
#define NVC 6
#define NHEADS 4

typedef __attribute__((ext_vector_type(16))) _Float16 v16h;
typedef __attribute__((ext_vector_type(8)))  float    v8f;

union AFrag { v16h v; uint4 q[2]; };

// ---------------- generic helpers ----------------
__global__ void fill_f32(float* p, float v, int n) {
  int i = blockIdx.x * blockDim.x + threadIdx.x;
  if (i < n) p[i] = v;
}

__device__ inline void atomicMaxF(float* addr, float val) {
  int old = __float_as_int(*addr);
  while (__int_as_float(old) < val) {
    int prev = atomicCAS((int*)addr, old, __float_as_int(val));
    if (prev == old) break;
    old = prev;
  }
}

// ---------------- edge list with self loops ----------------
__global__ void build_edges(const int* ei, int* src2, int* dst2) {
  int e = blockIdx.x * blockDim.x + threadIdx.x;
  if (e >= NET) return;
  if (e < NE) { src2[e] = ei[e]; dst2[e] = ei[NE + e]; }
  else { src2[e] = e - NE; dst2[e] = e - NE; }
}

// ---------------- h0 = x @ node_w + node_b ----------------
__global__ void node_embed(const float* __restrict__ x, const float* __restrict__ w,
                           const float* __restrict__ b, float* h, _Float16* h16) {
  int i = blockIdx.x * blockDim.x + threadIdx.x;
  if (i >= NN * HDIM) return;
  int n = i / HDIM, c = i % HDIM;
  float s = b[c];
#pragma unroll
  for (int j = 0; j < ND; j++) s += x[n * ND + j] * w[j * HDIM + c];
  h[i] = s; h16[i] = (_Float16)s;
}

// ---------------- ef = edge_attr @ ea_proj_w + b (f16 out) ----------------
__global__ void edge_proj(const float* __restrict__ ea, const float* __restrict__ w,
                          const float* __restrict__ b, _Float16* ef) {
  int i = blockIdx.x * blockDim.x + threadIdx.x;
  if (i >= NE * HDIM) return;
  int e = i / HDIM, c = i % HDIM;
  float s = b[c];
#pragma unroll
  for (int j = 0; j < ED; j++) s += ea[e * ED + j] * w[j * HDIM + c];
  ef[i] = (_Float16)s;
}

// ---------------- vnf = ctx@vnf_w+b ; c1 = att1_b + vnf @ att1_w[128:] ----------------
__global__ void vnf_c1(const float* __restrict__ vctx, const float* __restrict__ vw,
                       const float* __restrict__ vb, const float* __restrict__ a1w,
                       const float* __restrict__ a1b, float* c1) {
  __shared__ float sv[HDIM];
  int c = threadIdx.x;
  float s = vb[c];
#pragma unroll
  for (int j = 0; j < NVC; j++) s += vctx[j] * vw[j * HDIM + c];
  sv[c] = s;
  __syncthreads();
  float t = a1b[c];
  for (int h = 0; h < HDIM; h++) t += sv[h] * a1w[(HDIM + h) * HDIM + c];
  c1[c] = t;
}

// ---------------- transpose f32 W[K,NC] -> f16 Bt[NC,K] ----------------
__global__ void transpose_f16(const float* __restrict__ W, _Float16* Bt, int K, int NC) {
  int i = blockIdx.x * blockDim.x + threadIdx.x;
  if (i >= K * NC) return;
  int n = i / K, k = i % K;
  Bt[(size_t)n * K + k] = (_Float16)W[(size_t)k * NC + n];
}

// ---------------- We[l][h][k] = sum_c lew[l][h][k*128+c]*a_edge[l][k][c] ----------------
__global__ void compute_We(const float* __restrict__ lew, const float* __restrict__ aedge,
                           float* We) {
  int idx = blockIdx.x * blockDim.x + threadIdx.x;
  if (idx >= NL * HDIM * NHEADS) return;
  int l = idx / (HDIM * NHEADS);
  int r = idx % (HDIM * NHEADS);
  int h = r / NHEADS, k = r % NHEADS;
  const float* wrow = lew + ((size_t)l * HDIM + h) * (NHEADS * 128) + k * 128;
  const float* arow = aedge + ((size_t)l * NHEADS + k) * 128;
  float s = 0.f;
  for (int c = 0; c < 128; c++) s += wrow[c] * arow[c];
  We[idx] = s;
}

// ---------------- WMMA GEMM: C[M,NC] = act(A16[M,128] @ Bt16[NC,128] + bias) ----------------
template<int NC, bool RELU, bool HASB, bool OUTF16>
__global__ void wmma_gemm_k128(const _Float16* __restrict__ A,
                               const _Float16* __restrict__ Bt,
                               const float* __restrict__ bias,
                               float* __restrict__ Cf, _Float16* __restrict__ Ch, int M) {
  const int K = 128;
  int wave = (blockIdx.x * blockDim.x + threadIdx.x) >> 5;
  int lane = threadIdx.x & 31;
  const int ntiles = NC / 16;
  int mt = wave / ntiles;
  int nt = wave - mt * ntiles;
  if (mt * 16 >= M) return;          // uniform per wave: EXEC stays all-1s
  int hi = lane >> 4;                 // half-wave select
  int l16 = lane & 15;
  const _Float16* arow = A + (size_t)(mt * 16 + l16) * K;
  const _Float16* brow = Bt + (size_t)(nt * 16 + l16) * K;
  v8f acc = {};
#pragma unroll
  for (int kk = 0; kk < K; kk += 32) {
    AFrag a, b;
    // A 16x32 f16 layout: halves 0..7 <- K=kk+hi*8.. ; halves 8..15 <- K=kk+16+hi*8..
    a.q[0] = *(const uint4*)(arow + kk + hi * 8);
    a.q[1] = *(const uint4*)(arow + kk + 16 + hi * 8);
    // B 32x16 f16 layout (Bt is column-major of B): halves 0..7 <- K=kk+hi*8.. ; halves 8..15 <- K=kk+16+hi*8..
    b.q[0] = *(const uint4*)(brow + kk + hi * 8);
    b.q[1] = *(const uint4*)(brow + kk + 16 + hi * 8);
    acc = __builtin_amdgcn_wmma_f32_16x16x32_f16(false, a.v, false, b.v,
                                                 (short)0, acc, false, false);
  }
  int col = nt * 16 + l16;
  float bv = HASB ? bias[col] : 0.f;
#pragma unroll
  for (int r = 0; r < 8; r++) {
    int row = mt * 16 + hi * 8 + r;   // C layout: VGPR r -> M = r + (lane/16)*8
    float v = acc[r] + bv;
    if (RELU) v = fmaxf(v, 0.f);
    if (OUTF16) Ch[(size_t)row * NC + col] = (_Float16)v;
    else        Cf[(size_t)row * NC + col] = v;
  }
}

// ---------------- gate = sigmoid(a2 @ att3_w + b) ----------------
__global__ void gate_kernel(const _Float16* __restrict__ a2, const float* __restrict__ w3,
                            const float* __restrict__ b3, float* gate) {
  int e = blockIdx.x * blockDim.x + threadIdx.x;
  if (e >= NE) return;
  float s = b3[0];
  const _Float16* row = a2 + (size_t)e * 64;
#pragma unroll 8
  for (int j = 0; j < 64; j++) s += (float)row[j] * w3[j];
  gate[e] = 1.f / (1.f + __expf(-s));
}

// ---------------- e2[e][c] = ef[e][c]*gate[e] ----------------
__global__ void we_mul(const _Float16* __restrict__ ef, const float* __restrict__ gate,
                       _Float16* e2) {
  int i = blockIdx.x * blockDim.x + threadIdx.x;
  if (i >= NE * HDIM) return;
  e2[i] = (_Float16)((float)ef[i] * gate[i / HDIM]);
}

// ---------------- column mean of e2 rows < NE ----------------
__global__ void col_mean(const _Float16* __restrict__ e2, float* mean) {
  int c = blockIdx.x;
  float s = 0.f;
  for (int e = threadIdx.x; e < NE; e += blockDim.x)
    s += (float)e2[(size_t)e * HDIM + c];
  __shared__ float red[256];
  red[threadIdx.x] = s;
  __syncthreads();
  for (int off = 128; off > 0; off >>= 1) {
    if (threadIdx.x < off) red[threadIdx.x] += red[threadIdx.x + off];
    __syncthreads();
  }
  if (threadIdx.x == 0) mean[c] = red[0] / (float)NE;
}

__global__ void fill_self(const float* __restrict__ mean, _Float16* e2) {
  int i = blockIdx.x * blockDim.x + threadIdx.x;
  if (i >= NN * HDIM) return;
  e2[(size_t)NE * HDIM + i] = (_Float16)mean[i % HDIM];
}

// ---------------- al_e_all[l][e][k] = sum_h e2[e][h]*We[l][h][k] (all layers, one pass) ----------------
__global__ void compute_al_e(const _Float16* __restrict__ e2, const float* __restrict__ We,
                             float* ale) {
  __shared__ float sWe[NL * HDIM * NHEADS];
  for (int i = threadIdx.x; i < NL * HDIM * NHEADS; i += blockDim.x) sWe[i] = We[i];
  __syncthreads();
  int e = blockIdx.x * blockDim.x + threadIdx.x;
  if (e >= NET) return;
  float acc[NL][NHEADS] = {};
  const _Float16* row = e2 + (size_t)e * HDIM;
  for (int h = 0; h < HDIM; h++) {
    float v = (float)row[h];
    const float* wrow = &sWe[h * NHEADS];
#pragma unroll
    for (int l = 0; l < NL; l++)
#pragma unroll
      for (int k = 0; k < NHEADS; k++)
        acc[l][k] += v * wrow[l * HDIM * NHEADS + k];
  }
#pragma unroll
  for (int l = 0; l < NL; l++)
#pragma unroll
    for (int k = 0; k < NHEADS; k++)
      ale[((size_t)l * NET + e) * NHEADS + k] = acc[l][k];
}

// ---------------- al_s / al_d per node/head ----------------
__global__ void al_sd(const float* __restrict__ xh, const float* __restrict__ asrc,
                      const float* __restrict__ adst, float* als, float* ald, int l) {
  int i = blockIdx.x * blockDim.x + threadIdx.x;
  if (i >= NN * NHEADS) return;
  int n = i / NHEADS, k = i % NHEADS;
  const float* xr = xh + (size_t)n * 512 + k * 128;
  const float* as = asrc + ((size_t)l * NHEADS + k) * 128;
  const float* ad = adst + ((size_t)l * NHEADS + k) * 128;
  float s = 0.f, d = 0.f;
  for (int c = 0; c < 128; c++) { float v = xr[c]; s += v * as[c]; d += v * ad[c]; }
  als[i] = s; ald[i] = d;
}

// ---------------- alpha = leakyrelu(...) ; segment max via atomic CAS ----------------
__global__ void alpha_max(const float* __restrict__ als, const float* __restrict__ ald,
                          const float* __restrict__ ale, const int* __restrict__ src2,
                          const int* __restrict__ dst2, float* alpha, float* amax, int l) {
  int e = blockIdx.x * blockDim.x + threadIdx.x;
  if (e >= NET) return;
  int s = src2[e], d = dst2[e];
  const float* ar = ale + ((size_t)l * NET + e) * NHEADS;
#pragma unroll
  for (int k = 0; k < NHEADS; k++) {
    float a = als[s * NHEADS + k] + ald[d * NHEADS + k] + ar[k];
    a = a > 0.f ? a : 0.2f * a;
    alpha[e * NHEADS + k] = a;
    atomicMaxF(&amax[d * NHEADS + k], a);
  }
}

// ---------------- ex = exp(alpha - amax); den += ex ----------------
__global__ void exp_den(const int* __restrict__ dst2, const float* __restrict__ amax,
                        float* alpha, float* den) {
  int e = blockIdx.x * blockDim.x + threadIdx.x;
  if (e >= NET) return;
  int d = dst2[e];
#pragma unroll
  for (int k = 0; k < NHEADS; k++) {
    float ex = __expf(alpha[e * NHEADS + k] - amax[d * NHEADS + k]);
    alpha[e * NHEADS + k] = ex;
    atomicAdd(&den[d * NHEADS + k], ex);
  }
}

// ---------------- scatter: outacc[dst] += xh[src] * w ; one wave per edge ----------------
__global__ void gat_scatter(const float* __restrict__ xh, const float* __restrict__ exv,
                            const float* __restrict__ den, const int* __restrict__ src2,
                            const int* __restrict__ dst2, float* outacc) {
  int gid = blockIdx.x * blockDim.x + threadIdx.x;
  int e = gid >> 5, lane = gid & 31;
  if (e >= NET) return;
  int s = src2[e], d = dst2[e];
  float wk[NHEADS];
#pragma unroll
  for (int k = 0; k < NHEADS; k++)
    wk[k] = exv[e * NHEADS + k] / (den[d * NHEADS + k] + 1e-16f);
  const float* xr = xh + (size_t)s * 512;
  float* orow = outacc + (size_t)d * 512;
  __builtin_prefetch(xr, 0, 1);
#pragma unroll
  for (int i = lane; i < 512; i += 32)
    atomicAdd(&orow[i], xr[i] * wk[i >> 7]);
}

// ---------------- finalize: mean heads + bias, residual relu, LayerNorm (wave32) ----------------
__global__ void gat_finalize(const float* __restrict__ outacc, const float* __restrict__ gbias,
                             const float* __restrict__ lns, const float* __restrict__ lnb,
                             float* h, _Float16* h16, int l) {
  int gid = blockIdx.x * blockDim.x + threadIdx.x;
  int n = gid >> 5, lane = gid & 31;
  if (n >= NN) return;
  float t[4];
#pragma unroll
  for (int j = 0; j < 4; j++) {
    int c = lane + j * 32;
    const float* orow = outacc + (size_t)n * 512;
    float hc = 0.25f * (orow[c] + orow[128 + c] + orow[256 + c] + orow[384 + c])
             + gbias[l * HDIM + c];
    t[j] = h[(size_t)n * HDIM + c] + fmaxf(hc, 0.f);
  }
  float s = t[0] + t[1] + t[2] + t[3];
#pragma unroll
  for (int off = 16; off > 0; off >>= 1) s += __shfl_xor(s, off, 32);
  float m = s * (1.f / 128.f);
  float vs = 0.f;
#pragma unroll
  for (int j = 0; j < 4; j++) { float d = t[j] - m; vs += d * d; }
#pragma unroll
  for (int off = 16; off > 0; off >>= 1) vs += __shfl_xor(vs, off, 32);
  float inv = rsqrtf(vs * (1.f / 128.f) + 1e-5f);
#pragma unroll
  for (int j = 0; j < 4; j++) {
    int c = lane + j * 32;
    float o = (t[j] - m) * inv * lns[l * HDIM + c] + lnb[l * HDIM + c];
    h[(size_t)n * HDIM + c] = o;
    h16[(size_t)n * HDIM + c] = (_Float16)o;
  }
}

// ================= host side =================
static inline int nb(long long threads, int block) { return (int)((threads + block - 1) / block); }

extern "C" void kernel_launch(void* const* d_in, const int* in_sizes, int n_in,
                              void* d_out, int out_size, void* d_ws, size_t ws_size,
                              hipStream_t stream) {
  const float* x     = (const float*)d_in[0];
  const int*   ei    = (const int*)d_in[1];
  const float* eattr = (const float*)d_in[2];
  const float* vctx  = (const float*)d_in[3];
  const float* nodew = (const float*)d_in[4];
  const float* nodeb = (const float*)d_in[5];
  const float* eaw   = (const float*)d_in[6];
  const float* eab   = (const float*)d_in[7];
  const float* vnfw  = (const float*)d_in[8];
  const float* vnfb  = (const float*)d_in[9];
  const float* a1w   = (const float*)d_in[10];
  const float* a1b   = (const float*)d_in[11];
  const float* a2w   = (const float*)d_in[12];
  const float* a2b   = (const float*)d_in[13];
  const float* a3w   = (const float*)d_in[14];
  const float* a3b   = (const float*)d_in[15];
  const float* glw   = (const float*)d_in[16];
  const float* gas   = (const float*)d_in[17];
  const float* gad   = (const float*)d_in[18];
  const float* glew  = (const float*)d_in[19];
  const float* gae   = (const float*)d_in[20];
  const float* gbias = (const float*)d_in[21];
  const float* lns   = (const float*)d_in[22];
  const float* lnb   = (const float*)d_in[23];
  const float* outw  = (const float*)d_in[24];
  const float* outb  = (const float*)d_in[25];

  char* ws = (char*)d_ws;
  size_t off = 0;
  auto alloc = [&](size_t bytes) { size_t o = off; off = (off + bytes + 255) & ~(size_t)255; return o; };

  float*    h_f32  = (float*)   (ws + alloc((size_t)NN * HDIM * 4));
  _Float16* h_f16  = (_Float16*)(ws + alloc((size_t)NN * HDIM * 2));
  _Float16* ef16   = (_Float16*)(ws + alloc((size_t)NE * HDIM * 2));
  _Float16* a1_16  = (_Float16*)(ws + alloc((size_t)NE * HDIM * 2));
  _Float16* a2_16  = (_Float16*)(ws + alloc((size_t)NE * 64 * 2));
  float*    gate   = (float*)   (ws + alloc((size_t)NE * 4));
  _Float16* e2_16  = (_Float16*)(ws + alloc((size_t)NET * HDIM * 2));
  float*    mean   = (float*)   (ws + alloc(HDIM * 4));
  float*    c1     = (float*)   (ws + alloc(HDIM * 4));
  float*    We     = (float*)   (ws + alloc((size_t)NL * HDIM * NHEADS * 4));
  float*    ale    = (float*)   (ws + alloc((size_t)NL * NET * NHEADS * 4));
  float*    xh     = (float*)   (ws + alloc((size_t)NN * 512 * 4));
  float*    als    = (float*)   (ws + alloc((size_t)NN * NHEADS * 4));
  float*    ald    = (float*)   (ws + alloc((size_t)NN * NHEADS * 4));
  float*    alpha  = (float*)   (ws + alloc((size_t)NET * NHEADS * 4));
  float*    amax   = (float*)   (ws + alloc((size_t)NN * NHEADS * 4));
  float*    den    = (float*)   (ws + alloc((size_t)NN * NHEADS * 4));
  float*    outacc = (float*)   (ws + alloc((size_t)NN * 512 * 4));
  int*      src2   = (int*)     (ws + alloc((size_t)NET * 4));
  int*      dst2   = (int*)     (ws + alloc((size_t)NET * 4));
  _Float16* W1a_t  = (_Float16*)(ws + alloc((size_t)128 * 128 * 2));
  _Float16* W2t    = (_Float16*)(ws + alloc((size_t)64 * 128 * 2));
  _Float16* OutWt  = (_Float16*)(ws + alloc((size_t)256 * 128 * 2));
  _Float16* LWt    = (_Float16*)(ws + alloc((size_t)NL * 512 * 128 * 2));
  (void)ws_size; (void)in_sizes; (void)n_in; (void)out_size;

  const int B = 256;

  // ---- prep ----
  build_edges<<<nb(NET, B), B, 0, stream>>>(ei, src2, dst2);
  node_embed<<<nb((long long)NN * HDIM, B), B, 0, stream>>>(x, nodew, nodeb, h_f32, h_f16);
  edge_proj<<<nb((long long)NE * HDIM, B), B, 0, stream>>>(eattr, eaw, eab, ef16);
  vnf_c1<<<1, HDIM, 0, stream>>>(vctx, vnfw, vnfb, a1w, a1b, c1);
  transpose_f16<<<nb(128 * 128, B), B, 0, stream>>>(a1w, W1a_t, 128, 128);
  transpose_f16<<<nb(128 * 64, B), B, 0, stream>>>(a2w, W2t, 128, 64);
  transpose_f16<<<nb(128 * 256, B), B, 0, stream>>>(outw, OutWt, 128, 256);
  for (int l = 0; l < NL; l++)
    transpose_f16<<<nb(128 * 512, B), B, 0, stream>>>(glw + (size_t)l * 128 * 512,
                                                      LWt + (size_t)l * 512 * 128, 128, 512);
  compute_We<<<nb(NL * HDIM * NHEADS, B), B, 0, stream>>>(glew, gae, We);

  // ---- edge attention MLP (WMMA) ----
  wmma_gemm_k128<128, true, true, true><<<nb((long long)(NE / 16) * 8 * 32, B), B, 0, stream>>>(
      ef16, W1a_t, c1, nullptr, a1_16, NE);
  wmma_gemm_k128<64, true, true, true><<<nb((long long)(NE / 16) * 4 * 32, B), B, 0, stream>>>(
      a1_16, W2t, a2b, nullptr, a2_16, NE);
  gate_kernel<<<nb(NE, B), B, 0, stream>>>(a2_16, a3w, a3b, gate);
  we_mul<<<nb((long long)NE * HDIM, B), B, 0, stream>>>(ef16, gate, e2_16);
  col_mean<<<HDIM, 256, 0, stream>>>(e2_16, mean);
  fill_self<<<nb((long long)NN * HDIM, B), B, 0, stream>>>(mean, e2_16);
  compute_al_e<<<nb(NET, B), B, 0, stream>>>(e2_16, We, ale);

  // ---- GAT layers ----
  for (int l = 0; l < NL; l++) {
    wmma_gemm_k128<512, false, false, false><<<nb((long long)(NN / 16) * 32 * 32, B), B, 0, stream>>>(
        h_f16, LWt + (size_t)l * 512 * 128, nullptr, xh, nullptr, NN);
    al_sd<<<nb(NN * NHEADS, B), B, 0, stream>>>(xh, gas, gad, als, ald, l);
    fill_f32<<<nb(NN * NHEADS, B), B, 0, stream>>>(amax, -1e30f, NN * NHEADS);
    fill_f32<<<nb(NN * NHEADS, B), B, 0, stream>>>(den, 0.f, NN * NHEADS);
    fill_f32<<<nb((long long)NN * 512, B), B, 0, stream>>>(outacc, 0.f, NN * 512);
    alpha_max<<<nb(NET, B), B, 0, stream>>>(als, ald, ale, src2, dst2, alpha, amax, l);
    exp_den<<<nb(NET, B), B, 0, stream>>>(dst2, amax, alpha, den);
    gat_scatter<<<nb((long long)NET * 32, B), B, 0, stream>>>(xh, alpha, den, src2, dst2, outacc);
    gat_finalize<<<nb((long long)NN * 32, B), B, 0, stream>>>(outacc, gbias, lns, lnb, h_f32, h_f16, l);
  }

  // ---- output projection (WMMA) ----
  wmma_gemm_k128<256, false, true, false><<<nb((long long)(NN / 16) * 16 * 32, B), B, 0, stream>>>(
      h_f16, OutWt, outb, (float*)d_out, nullptr, NN);
}
